// Block_72756745994545
// MI455X (gfx1250) — matrix operand order repaired and checked by reading, not compiled, not run on hardware
//
#include <hip/hip_runtime.h>
#include <hip/hip_bf16.h>

// ---------------------------------------------------------------------------
// RWKV-4 layer forward for MI455X (gfx1250, wave32, WMMA).
// All 7 einsums are X[M,K] · W[N,K]^T GEMMs -> register-blocked WMMA kernel:
// each wave computes a 32x64 strip (2x4 grid of 16x16 v_wmma_f32_16x16x32_f16
// tiles, 8 WMMAs per K=32 step), 8 waves/block cover a 128x128 block tile.
// GEMM loads use addrspace(1) byte-offset addressing (uniform base + 32-bit
// lane offsets) so the backend emits global_load_b128 saddr form with no
// per-load 64-bit VALU address math. K loop unrolled x2 to give the scheduler
// disjoint register sets (fewer WMMA->load WAR hazard NOPs).
// WKV recurrence is a per-(b,c)-channel sequential scan (4096 channels).
// ---------------------------------------------------------------------------

typedef _Float16 half_t;
typedef __attribute__((ext_vector_type(16))) _Float16 v16h;
typedef __attribute__((ext_vector_type(8)))  _Float16 v8h;
typedef __attribute__((ext_vector_type(8)))  float    v8f;

// Global (addrspace 1) pointer types for the GEMM hot path.
typedef const __attribute__((address_space(1))) unsigned char* gcb_p;
typedef const __attribute__((address_space(1))) v8h*    gv8h_p;
typedef const __attribute__((address_space(1))) v16h*   gv16h_p;
typedef const __attribute__((address_space(1))) float*  gcf_p;
typedef __attribute__((address_space(1))) float*        gf_p;
typedef __attribute__((address_space(1))) half_t*       gh_p;

#define BDIM 256
#define WAVES_PER_BLOCK 8   // 256 threads / wave32

// Per-wave register tile: 2 (M) x 4 (N) WMMA tiles = 32 x 64 outputs.
#define WTM 2
#define WTN 4
// Block tile: 4 waves in M, 2 waves in N -> 128 x 128.
#define BLK_M (16 * WTM * 4)   // 128
#define BLK_N (16 * WTN * 2)   // 128

// -------------------------- elementwise kernels ----------------------------

__global__ void f32_to_f16_kernel(const float* __restrict__ in,
                                  half_t* __restrict__ out, size_t n) {
  size_t i = (size_t)blockIdx.x * blockDim.x + threadIdx.x;
  if (i < n) out[i] = (half_t)in[i];
}

// LayerNorm, one block per row of C elements.
__global__ __launch_bounds__(BDIM) void ln_kernel(
    const float* __restrict__ x, const float* __restrict__ g,
    const float* __restrict__ b, float* __restrict__ h, int C) {
  __shared__ float red[BDIM];
  const size_t row = blockIdx.x;
  const float* xr = x + row * (size_t)C;

  float s = 0.f;
  for (int c = threadIdx.x; c < C; c += BDIM) s += xr[c];
  red[threadIdx.x] = s;
  __syncthreads();
  for (int o = BDIM / 2; o > 0; o >>= 1) {
    if ((int)threadIdx.x < o) red[threadIdx.x] += red[threadIdx.x + o];
    __syncthreads();
  }
  const float mean = red[0] / (float)C;
  __syncthreads();

  float vs = 0.f;
  for (int c = threadIdx.x; c < C; c += BDIM) {
    float d = xr[c] - mean;
    vs += d * d;
  }
  red[threadIdx.x] = vs;
  __syncthreads();
  for (int o = BDIM / 2; o > 0; o >>= 1) {
    if ((int)threadIdx.x < o) red[threadIdx.x] += red[threadIdx.x + o];
    __syncthreads();
  }
  const float rstd = rsqrtf(red[0] / (float)C + 1e-5f);

  float* hr = h + row * (size_t)C;
  for (int c = threadIdx.x; c < C; c += BDIM)
    hr[c] = (xr[c] - mean) * rstd * g[c] + b[c];
}

// Token-shift mix: out = h*t + shift(h)*(1-t), converted to f16 for WMMA.
// xv/tv may be null (channel-mix path needs only xk/xr).
__global__ void mix_kernel(const float* __restrict__ h,
                           const float* __restrict__ tk,
                           const float* __restrict__ tv,
                           const float* __restrict__ tr,
                           half_t* __restrict__ xk, half_t* __restrict__ xv,
                           half_t* __restrict__ xr,
                           int T, int C, size_t total) {
  size_t i = (size_t)blockIdx.x * blockDim.x + threadIdx.x;
  if (i >= total) return;
  const int    c = (int)(i % C);
  const size_t m = i / C;
  const int    t = (int)(m % (size_t)T);
  const float hc = h[i];
  const float hp = (t == 0) ? 0.f : h[i - C];  // shift: previous token, 0 at t==0
  float a;
  a = tk[c]; xk[i] = (half_t)(hc * a + hp * (1.f - a));
  if (xv) { a = tv[c]; xv[i] = (half_t)(hc * a + hp * (1.f - a)); }
  a = tr[c]; xr[i] = (half_t)(hc * a + hp * (1.f - a));
}

// rwkv = sigmoid(r) * wkv, to f16 for the output projection GEMM.
__global__ void gate_kernel(const float* __restrict__ sr,
                            const float* __restrict__ wkv,
                            half_t* __restrict__ out, size_t n) {
  size_t i = (size_t)blockIdx.x * blockDim.x + threadIdx.x;
  if (i < n) out[i] = (half_t)(sr[i] * wkv[i]);
}

// final: out = x1 + sigmoid(r2) * kv   (sigmoid already fused in GEMM epilogue)
__global__ void final_kernel(const float* __restrict__ x1,
                             const float* __restrict__ r2,
                             const float* __restrict__ kv,
                             float* __restrict__ out, size_t n) {
  size_t i = (size_t)blockIdx.x * blockDim.x + threadIdx.x;
  if (i < n) out[i] = x1[i] + r2[i] * kv[i];
}

// ------------------------------ WKV scan -----------------------------------
// Numerically-stable RWKV-4 recurrence; one thread per (b, c) channel.
__global__ void wkv_kernel(const float* __restrict__ td,
                           const float* __restrict__ tf,
                           const float* __restrict__ k,
                           const float* __restrict__ v,
                           float* __restrict__ wkv,
                           int Bn, int T, int C) {
  const int gid = blockIdx.x * blockDim.x + threadIdx.x;
  if (gid >= Bn * C) return;
  const int b = gid / C;
  const int c = gid % C;
  const float w = -__expf(td[c]);
  const float u = tf[c];
  float aa = 0.f, bb = 0.f, pp = -1e38f;
  size_t idx = (size_t)b * T * C + c;
  for (int t = 0; t < T; ++t, idx += C) {
    const float kk = k[idx];
    const float vv = v[idx];
    const float ww = u + kk;
    const float p  = fmaxf(pp, ww);
    const float e1 = __expf(pp - p);
    const float e2 = __expf(ww - p);
    wkv[idx] = (e1 * aa + e2 * vv) / (e1 * bb + e2);
    const float ww2 = pp + w;
    const float p2  = fmaxf(ww2, kk);
    const float f1  = __expf(ww2 - p2);
    const float f2  = __expf(kk - p2);
    aa = f1 * aa + f2 * vv;
    bb = f1 * bb + f2;
    pp = p2;
  }
}

// ------------------------------ WMMA GEMM ----------------------------------
// out[M,N] = A[M,K](f16, row-major) x W[N,K](f16, row-major)^T, f32 accum.
// Register-blocked: each wave owns a 32x64 strip = WTM x WTN 16x16 WMMA tiles.
// Per K=32 step: 2 A fragments + 4 B fragments feed 8 v_wmma ops
// (~21 FLOP per loaded byte per wave, before L0/L2 dedup across waves).
//
// Fragment layouts used here (16-bit, wave32; A/B use contiguous k, not the ISA table k order):
//   A (16x32): lane m = lane&15; 16 contiguous K halves starting at
//              k0 + 16*(lane>>4) of A row m (matches B).
//   B (32x16): col n = lane&15; 16 contiguous K halves starting at
//              k0 + 16*(lane>>4) of W row n.
//   C/D (16x16 f32): lanes 0..15 -> M=vgpr, lanes 16..31 -> M=8+vgpr; N=lane&15.
enum { EP_NONE = 0, EP_SIGMOID = 1, EP_RESID = 2, EP_RELU2_F16 = 3 };

template <int EP>
__global__ __launch_bounds__(BDIM) void gemm_wmma_kernel(
    const half_t* __restrict__ A, const half_t* __restrict__ W,
    const float* __restrict__ resid, float* __restrict__ outF,
    half_t* __restrict__ outH, int M, int N, int K) {
  const int lane  = threadIdx.x & 31;
  const int wave  = threadIdx.x >> 5;
  const int waveM = wave >> 1;      // 0..3
  const int waveN = wave & 1;       // 0..1

  const int mBase = blockIdx.y * BLK_M + waveM * (16 * WTM);  // 32-row strip
  const int nBase = blockIdx.x * BLK_N + waveN * (16 * WTN);  // 64-col strip

  const int kAoff = (lane >> 4) * 16;
  const int kBoff = (lane >> 4) * 16;
  const int lmod  = lane & 15;

  // Global (AS1) byte-addressed bases: uniform SGPR base + 32-bit byte
  // offsets -> global_load_b128 saddr form, no 64-bit VALU address math.
  gcb_p Ab = (gcb_p)A;
  gcb_p Wb = (gcb_p)W;

  unsigned aB[WTM];   // byte offset of each A fragment row at K=0
#pragma unroll
  for (int i = 0; i < WTM; ++i)
    aB[i] = ((unsigned)(mBase + i * 16 + lmod) * (unsigned)K +
             (unsigned)kAoff) * 2u;
  unsigned bB[WTN];   // byte offset of each B fragment row at K=0
#pragma unroll
  for (int j = 0; j < WTN; ++j)
    bB[j] = ((unsigned)(nBase + j * 16 + lmod) * (unsigned)K +
             (unsigned)kBoff) * 2u;

  v8f acc[WTM][WTN];
#pragma unroll
  for (int i = 0; i < WTM; ++i)
#pragma unroll
    for (int j = 0; j < WTN; ++j) acc[i][j] = (v8f){};

  const unsigned Kbytes = (unsigned)K * 2u;  // bytes per row; K step = 64B
#pragma unroll 2
  for (unsigned kb = 0; kb < Kbytes; kb += 64u) {
    union { v16h v; v8h h8[2]; } afr[WTM];
#pragma unroll
    for (int i = 0; i < WTM; ++i) {
      afr[i].h8[0] = *(gv8h_p)(Ab + (aB[i] + kb));         // K = +kAoff..+8
      afr[i].h8[1] = *(gv8h_p)(Ab + (aB[i] + kb + 16u));   // K = +8+kAoff..+8
    }
    v16h bfr[WTN];
#pragma unroll
    for (int j = 0; j < WTN; ++j)
      bfr[j] = *(gv16h_p)(Wb + (bB[j] + kb));              // 16 contiguous K
#pragma unroll
    for (int i = 0; i < WTM; ++i)
#pragma unroll
      for (int j = 0; j < WTN; ++j)
        acc[i][j] = __builtin_amdgcn_wmma_f32_16x16x32_f16(
            /*neg_a=*/false, afr[i].v, /*neg_b=*/false, bfr[j],
            /*c_mod=*/(short)0, acc[i][j],
            /*reuse_a=*/false, /*reuse_b=*/false);
  }

  // Epilogue: write WTM x WTN 16x16 tiles through the global path.
  gf_p  outFg   = (gf_p)outF;
  gh_p  outHg   = (gh_p)outH;
  gcf_p residg  = (gcf_p)resid;
  const int mHalf = (lane >> 4) * 8;
#pragma unroll
  for (int i = 0; i < WTM; ++i) {
    const int mT = mBase + i * 16 + mHalf;
#pragma unroll
    for (int j = 0; j < WTN; ++j) {
      const int n = nBase + j * 16 + lmod;
#pragma unroll
      for (int r = 0; r < 8; ++r) {
        const unsigned o = (unsigned)(mT + r) * (unsigned)N + (unsigned)n;
        const float val = acc[i][j][r];
        if (EP == EP_NONE) {
          outFg[o] = val;
        } else if (EP == EP_SIGMOID) {
          outFg[o] = 1.f / (1.f + __expf(-val));
        } else if (EP == EP_RESID) {
          outFg[o] = residg[o] + val;
        } else {  // EP_RELU2_F16
          const float rr = val > 0.f ? val : 0.f;
          outHg[o] = (half_t)(rr * rr);
        }
      }
    }
  }
}

// ------------------------------- launch ------------------------------------

static inline unsigned gridFor(size_t n) {
  return (unsigned)((n + BDIM - 1) / BDIM);
}

extern "C" void kernel_launch(void* const* d_in, const int* in_sizes, int n_in,
                              void* d_out, int out_size, void* d_ws, size_t ws_size,
                              hipStream_t stream) {
  (void)in_sizes; (void)n_in; (void)out_size; (void)ws_size;

  constexpr int Bn = 4, T = 2048, C = 1024, FFN = 4096;
  constexpr int M = Bn * T;  // 8192 token rows

  // Inputs (setup_inputs order, all f32)
  const float* x        = (const float*)d_in[0];
  const float* ln1_w    = (const float*)d_in[1];
  const float* ln1_b    = (const float*)d_in[2];
  const float* ln2_w    = (const float*)d_in[3];
  const float* ln2_b    = (const float*)d_in[4];
  const float* t_decay  = (const float*)d_in[5];
  const float* t_first  = (const float*)d_in[6];
  const float* tmk      = (const float*)d_in[7];
  const float* tmv      = (const float*)d_in[8];
  const float* tmr      = (const float*)d_in[9];
  const float* att_kw   = (const float*)d_in[10];
  const float* att_vw   = (const float*)d_in[11];
  const float* att_rw   = (const float*)d_in[12];
  const float* att_ow   = (const float*)d_in[13];
  const float* f_tmk    = (const float*)d_in[14];
  const float* f_tmr    = (const float*)d_in[15];
  const float* f_kw     = (const float*)d_in[16];  // [FFN, C]
  const float* f_rw     = (const float*)d_in[17];  // [C, C]
  const float* f_vw     = (const float*)d_in[18];  // [C, FFN]
  float* out = (float*)d_out;

  const size_t MC = (size_t)M * C;
  const size_t CC = (size_t)C * C;
  const size_t FC = (size_t)FFN * C;

  // ---- workspace layout (phase-2 buffers alias dead phase-1 buffers) ----
  char* ws = (char*)d_ws;
  size_t off = 0;
  auto alloc = [&](size_t bytes) {
    char* p = ws + off;
    off += (bytes + 255) & ~(size_t)255;
    return p;
  };
  float*  h      = (float*) alloc(MC * 4);   // LN output (both phases)
  half_t* xk16   = (half_t*)alloc(MC * 2);
  half_t* xv16   = (half_t*)alloc(MC * 2);
  half_t* xr16   = (half_t*)alloc(MC * 2);
  half_t* wk16   = (half_t*)alloc(CC * 2);
  half_t* wv16   = (half_t*)alloc(CC * 2);
  half_t* wr16   = (half_t*)alloc(CC * 2);
  half_t* wo16   = (half_t*)alloc(CC * 2);
  half_t* fkw16  = (half_t*)alloc(FC * 2);
  half_t* fvw16  = (half_t*)alloc(FC * 2);
  half_t* frw16  = (half_t*)alloc(CC * 2);
  float*  kbuf   = (float*) alloc(MC * 4);
  float*  vbuf   = (float*) alloc(MC * 4);   // contiguous with wkvb...
  float*  wkvb   = (float*) alloc(MC * 4);   // ...so kk16 (M*FFN*2) overlays both
  float*  srb    = (float*) alloc(MC * 4);
  half_t* rwkv16 = (half_t*)alloc(MC * 2);
  float*  x1     = (float*) alloc(MC * 4);
  // phase-2 aliases over dead phase-1 storage:
  half_t* kk16 = (half_t*)vbuf;  // M*FFN f16 = 64MB over vbuf+wkvb (32+32MB)
  float*  kvb  = kbuf;
  float*  r2b  = srb;

  // ---- weight conversion f32 -> f16 ----
  f32_to_f16_kernel<<<gridFor(CC), BDIM, 0, stream>>>(att_kw, wk16, CC);
  f32_to_f16_kernel<<<gridFor(CC), BDIM, 0, stream>>>(att_vw, wv16, CC);
  f32_to_f16_kernel<<<gridFor(CC), BDIM, 0, stream>>>(att_rw, wr16, CC);
  f32_to_f16_kernel<<<gridFor(CC), BDIM, 0, stream>>>(att_ow, wo16, CC);
  f32_to_f16_kernel<<<gridFor(FC), BDIM, 0, stream>>>(f_kw,  fkw16, FC);
  f32_to_f16_kernel<<<gridFor(FC), BDIM, 0, stream>>>(f_vw,  fvw16, FC);
  f32_to_f16_kernel<<<gridFor(CC), BDIM, 0, stream>>>(f_rw,  frw16, CC);

  auto gemmGrid = [&](int mM, int nN) {
    return dim3((unsigned)(nN / BLK_N), (unsigned)(mM / BLK_M));
  };

  // ---------------- Phase 1: TimeMix ----------------
  ln_kernel<<<M, BDIM, 0, stream>>>(x, ln1_w, ln1_b, h, C);
  mix_kernel<<<gridFor(MC), BDIM, 0, stream>>>(h, tmk, tmv, tmr,
                                               xk16, xv16, xr16, T, C, MC);
  gemm_wmma_kernel<EP_NONE><<<gemmGrid(M, C), BDIM, 0, stream>>>(
      xk16, wk16, nullptr, kbuf, nullptr, M, C, C);
  gemm_wmma_kernel<EP_NONE><<<gemmGrid(M, C), BDIM, 0, stream>>>(
      xv16, wv16, nullptr, vbuf, nullptr, M, C, C);
  gemm_wmma_kernel<EP_SIGMOID><<<gemmGrid(M, C), BDIM, 0, stream>>>(
      xr16, wr16, nullptr, srb, nullptr, M, C, C);
  wkv_kernel<<<(Bn * C + BDIM - 1) / BDIM, BDIM, 0, stream>>>(
      t_decay, t_first, kbuf, vbuf, wkvb, Bn, T, C);
  gate_kernel<<<gridFor(MC), BDIM, 0, stream>>>(srb, wkvb, rwkv16, MC);
  gemm_wmma_kernel<EP_RESID><<<gemmGrid(M, C), BDIM, 0, stream>>>(
      rwkv16, wo16, x, x1, nullptr, M, C, C);

  // ---------------- Phase 2: ChannelMix ----------------
  ln_kernel<<<M, BDIM, 0, stream>>>(x1, ln2_w, ln2_b, h, C);
  mix_kernel<<<gridFor(MC), BDIM, 0, stream>>>(h, f_tmk, nullptr, f_tmr,
                                               xk16, nullptr, xr16, T, C, MC);
  gemm_wmma_kernel<EP_RELU2_F16><<<gemmGrid(M, FFN), BDIM, 0, stream>>>(
      xk16, fkw16, nullptr, nullptr, kk16, M, FFN, C);
  gemm_wmma_kernel<EP_NONE><<<gemmGrid(M, C), BDIM, 0, stream>>>(
      kk16, fvw16, nullptr, kvb, nullptr, M, C, FFN);
  gemm_wmma_kernel<EP_SIGMOID><<<gemmGrid(M, C), BDIM, 0, stream>>>(
      xr16, frw16, nullptr, r2b, nullptr, M, C, C);
  final_kernel<<<gridFor(MC), BDIM, 0, stream>>>(x1, r2b, kvb, out, MC);
}
